// GAT_59742995087819
// MI455X (gfx1250) — matrix operand order repaired and checked by reading, not compiled, not run on hardware
//
#include <hip/hip_runtime.h>
#include <hip/hip_bf16.h>

// Problem constants
#define BB   16
#define NN   1024
#define NF   128
#define NL   3
#define HH   64      // AGG_HIDDEN
#define FCH  128
#define NC   10
#define ATT_LD (NN + 8)   // padded f16 row stride in LDS (keeps 16B alignment: 1032*2=2064=129*16)

typedef __attribute__((ext_vector_type(16))) _Float16 v16h;
typedef __attribute__((ext_vector_type(8)))  float    v8f;

union Frag16 { v16h v; uint4 q[2]; };

// A fragment (16x32 f16): lane holds row (lane&15); 16 contiguous K values starting at k0 + half*16.
__device__ __forceinline__ v16h load_a_frag(const _Float16* row_ptr, int k0, int half) {
  Frag16 f;
  const _Float16* p = row_ptr + k0 + half * 16;
  f.q[0] = *(const uint4*)(p);
  f.q[1] = *(const uint4*)(p + 8);
  return f.v;
}

// B fragment (32x16 f16) from a TRANSPOSED operand [N][K]: lane holds column (lane&15),
// 16 contiguous K values starting at k0 + (lane>>4)*16.
__device__ __forceinline__ v16h load_b_frag(const _Float16* col_row_ptr, int kb) {
  Frag16 f;
  f.q[0] = *(const uint4*)(col_row_ptr + kb);
  f.q[1] = *(const uint4*)(col_row_ptr + kb + 8);
  return f.v;
}

__device__ __forceinline__ v8f wmma_f16(v16h a, v16h b, v8f c) {
  return __builtin_amdgcn_wmma_f32_16x16x32_f16(false, a, false, b, (short)0, c, false, false);
}

// ---------------------------------------------------------------------------
// Kernel 1: convert x -> f16, W -> W^T f16, zero pooled accumulator
// ---------------------------------------------------------------------------
__global__ void prep_kernel(const float* __restrict__ x, const float* __restrict__ Ws,
                            _Float16* __restrict__ xh, _Float16* __restrict__ WT,
                            float* __restrict__ pooled) {
  int t = blockIdx.x * blockDim.x + threadIdx.x;
  int stride = gridDim.x * blockDim.x;
  for (int i = t; i < BB * NN * NF; i += stride) xh[i] = (_Float16)x[i];
  // WT[l][c][k] = Ws[l][k][c]
  for (int i = t; i < NL * HH * NF; i += stride) {
    int l = i / (HH * NF);
    int rem = i - l * HH * NF;
    int c = rem >> 7;          // /NF
    int k = rem & (NF - 1);
    WT[i] = (_Float16)Ws[(l * NF + k) * HH + c];
  }
  for (int i = t; i < BB * NL * HH; i += stride) pooled[i] = 0.0f;
}

// ---------------------------------------------------------------------------
// Kernel 2: h = x @ W[l] via WMMA. Emits hT (f16, [l][b][64][1024]) and
// f1/f2 (f32) computed in-register from the f32 accumulators.
// grid = (128 row-blocks, 3 layers), block = 256 (8 waves x 16 rows)
// ---------------------------------------------------------------------------
__global__ __launch_bounds__(256) void gemm1_kernel(
    const _Float16* __restrict__ xh, const _Float16* __restrict__ WT,
    const float* __restrict__ a1, const float* __restrict__ a2,
    _Float16* __restrict__ hT, float* __restrict__ f1, float* __restrict__ f2) {
  const int l = blockIdx.y;
  const int w = threadIdx.x >> 5;
  const int lane = threadIdx.x & 31;
  const int cl = lane & 15;
  const int half = lane >> 4;
  const int rowbase = blockIdx.x * 128 + w * 16;

  const _Float16* arow = xh + (size_t)(rowbase + cl) * NF;

  v8f acc[4] = {};
#pragma unroll
  for (int s = 0; s < 4; ++s) {          // K = 128 in steps of 32
    const int k0 = s * 32;
    v16h A = load_a_frag(arow, k0, half);
#pragma unroll
    for (int cg = 0; cg < 4; ++cg) {     // N = 64 in 16-col groups
      const _Float16* bp = WT + (size_t)(l * HH + cg * 16 + cl) * NF;
      v16h Bf = load_b_frag(bp, k0 + half * 16);
      acc[cg] = wmma_f16(A, Bf, acc[cg]);
    }
  }

  const int bb = rowbase >> 10;          // batch
  const int nblk = rowbase & (NN - 1);   // node offset in batch
  const int base = half * 8;             // row half within tile

  // hT[l][bb][c][n] stores: per (cg), 8 consecutive f16 along n -> one 16B store
#pragma unroll
  for (int cg = 0; cg < 4; ++cg) {
    union { uint4 q; _Float16 h[8]; } u;
#pragma unroll
    for (int r = 0; r < 8; ++r) u.h[r] = (_Float16)acc[cg][r];
    size_t addr = ((size_t)(l * BB + bb) * HH + cg * 16 + cl) * NN + nblk + base;
    *(uint4*)(hT + addr) = u.q;
  }

  // f1 = h . a1, f2 = h . a2 per row (cross-lane reduce within 16-lane halves)
  float a1c[4], a2c[4];
#pragma unroll
  for (int cg = 0; cg < 4; ++cg) {
    a1c[cg] = a1[l * HH + cg * 16 + cl];
    a2c[cg] = a2[l * HH + cg * 16 + cl];
  }
#pragma unroll
  for (int r = 0; r < 8; ++r) {
    float s1 = 0.f, s2 = 0.f;
#pragma unroll
    for (int cg = 0; cg < 4; ++cg) {
      float hv = acc[cg][r];
      s1 = fmaf(hv, a1c[cg], s1);
      s2 = fmaf(hv, a2c[cg], s2);
    }
#pragma unroll
    for (int m = 8; m; m >>= 1) {
      s1 += __shfl_xor(s1, m, 16);
      s2 += __shfl_xor(s2, m, 16);
    }
    if (cl == 0) {
      int n = nblk + base + r;
      f1[(l * BB + bb) * NN + n] = s1;
      f2[(l * BB + bb) * NN + n] = s2;
    }
  }
}

// ---------------------------------------------------------------------------
// Kernel 3: fused attention for all 3 layers. One block per (b, 16-row block).
// adj is streamed once per layer (NL times per element). att @ h done with WMMA, K split
// across the 8 waves, cross-wave reduce in LDS, relu + mean-pool accumulate.
// ---------------------------------------------------------------------------
__global__ __launch_bounds__(256) void attn_kernel(
    const float* __restrict__ adj, const _Float16* __restrict__ hT,
    const float* __restrict__ f1, const float* __restrict__ f2,
    float* __restrict__ pooled) {
  __shared__ float    s_sc[16 * NN];          // 64KB: scores, then reused as partials
  __shared__ _Float16 s_att[16 * ATT_LD];     // 33KB: unnormalized exp weights (f16)
  __shared__ float    s_f2[NN];               // 4KB
  __shared__ float    s_inv[16];
  __shared__ float    s_pool[NL * HH];
  float* s_part = s_sc;                       // alias: scores dead during reduce

  const int b = blockIdx.y;
  const int ib = blockIdx.x * 16;
  const int tid = threadIdx.x;
  const int w = tid >> 5;
  const int lane = tid & 31;
  const int cl = lane & 15;
  const int half = lane >> 4;

  for (int i = tid; i < NL * HH; i += 256) s_pool[i] = 0.f;

  for (int l = 0; l < NL; ++l) {
    __syncthreads();  // protects s_sc/s_att/s_f2 reuse across layers
    for (int j = tid; j < NN; j += 256) s_f2[j] = f2[(l * BB + b) * NN + j];
    __syncthreads();

    // ---- masked leaky scores + online softmax stats (wave w owns rows 2w, 2w+1)
    for (int rr = 0; rr < 2; ++rr) {
      const int i = 2 * w + rr;
      const float f1v = f1[(l * BB + b) * NN + ib + i];
      const float* arow = adj + ((size_t)b * NN + ib + i) * NN;
      float mx = -3.4e38f;
      for (int j = lane; j < NN; j += 32) {
        float e = f1v + s_f2[j];
        e = e > 0.f ? e : 0.2f * e;                 // leaky_relu(0.2)
        float s = (arow[j] > 0.f) ? e : -9.0e15f;   // NEG_INF mask
        s_sc[i * NN + j] = s;
        mx = fmaxf(mx, s);
      }
#pragma unroll
      for (int m = 16; m; m >>= 1) mx = fmaxf(mx, __shfl_xor(mx, m));
      float sum = 0.f;
      for (int j = lane; j < NN; j += 32) {
        float p = __expf(s_sc[i * NN + j] - mx);
        sum += p;
        s_att[i * ATT_LD + j] = (_Float16)p;        // unnormalized; 1/sum folded later
      }
#pragma unroll
      for (int m = 16; m; m >>= 1) sum += __shfl_xor(sum, m);
      if (lane == 0) s_inv[i] = 1.f / sum;
    }
    __syncthreads();

    // ---- out[16x64] = att16[16x1024] @ h[1024x64] (B from hT), K split over waves
    v8f acc[4] = {};
    const _Float16* att_row = s_att + (size_t)cl * ATT_LD;
#pragma unroll
    for (int s = 0; s < 4; ++s) {
      const int k0 = w * 128 + s * 32;
      v16h A = load_a_frag(att_row, k0, half);      // ds_load_b128 x2
#pragma unroll
      for (int cg = 0; cg < 4; ++cg) {
        const _Float16* bp = hT + ((size_t)(l * BB + b) * HH + cg * 16 + cl) * NN;
        v16h Bf = load_b_frag(bp, k0 + half * 16);
        acc[cg] = wmma_f16(A, Bf, acc[cg]);
      }
    }
    // per-wave partial C -> LDS
#pragma unroll
    for (int cg = 0; cg < 4; ++cg)
#pragma unroll
      for (int r = 0; r < 8; ++r)
        s_part[(w * 16 + half * 8 + r) * HH + cg * 16 + cl] = acc[cg][r];
    __syncthreads();

    // ---- cross-wave reduce, scale by 1/sum, relu, pool-accumulate
    for (int idx = tid; idx < 16 * HH; idx += 256) {
      const int mm = idx >> 6;
      const int cc = idx & (HH - 1);
      float v = 0.f;
#pragma unroll
      for (int ww = 0; ww < 8; ++ww) v += s_part[(ww * 16 + mm) * HH + cc];
      v *= s_inv[mm];
      v = fmaxf(v, 0.f);
      atomicAdd(&s_pool[l * HH + cc], v);
    }
  }
  __syncthreads();
  for (int i = tid; i < NL * HH; i += 256)
    atomicAdd(&pooled[b * (NL * HH) + i], s_pool[i] * (1.0f / (float)NN));
}

// ---------------------------------------------------------------------------
// Kernel 4: MLP head + softmax (tiny: one block)
// ---------------------------------------------------------------------------
__global__ __launch_bounds__(256) void final_kernel(
    const float* __restrict__ pooled, const float* __restrict__ W1,
    const float* __restrict__ b1, const float* __restrict__ W2,
    const float* __restrict__ b2, float* __restrict__ out) {
  __shared__ float s_p[BB * NL * HH];
  __shared__ float s_z[BB * FCH];
  __shared__ float s_logit[BB * NC];
  const int t = threadIdx.x;
  for (int i = t; i < BB * NL * HH; i += 256) s_p[i] = pooled[i];
  __syncthreads();
  for (int i = t; i < BB * FCH; i += 256) {
    const int bb = i >> 7, j = i & (FCH - 1);
    float acc = b1[j];
    for (int k = 0; k < NL * HH; ++k) acc = fmaf(s_p[bb * NL * HH + k], W1[k * FCH + j], acc);
    s_z[i] = fmaxf(acc, 0.f);
  }
  __syncthreads();
  for (int i = t; i < BB * NC; i += 256) {
    const int bb = i / NC, c = i - bb * NC;
    float acc = b2[c];
    for (int j = 0; j < FCH; ++j) acc = fmaf(s_z[bb * FCH + j], W2[j * NC + c], acc);
    s_logit[i] = acc;
  }
  __syncthreads();
  if (t < BB) {
    float mx = -3.4e38f;
    for (int c = 0; c < NC; ++c) mx = fmaxf(mx, s_logit[t * NC + c]);
    float e[NC], sm = 0.f;
    for (int c = 0; c < NC; ++c) { e[c] = __expf(s_logit[t * NC + c] - mx); sm += e[c]; }
    const float inv = 1.f / sm;
    for (int c = 0; c < NC; ++c) out[t * NC + c] = e[c] * inv;
  }
}

// ---------------------------------------------------------------------------
extern "C" void kernel_launch(void* const* d_in, const int* in_sizes, int n_in,
                              void* d_out, int out_size, void* d_ws, size_t ws_size,
                              hipStream_t stream) {
  (void)in_sizes; (void)n_in; (void)out_size; (void)ws_size;
  const float* x   = (const float*)d_in[0];
  const float* adj = (const float*)d_in[1];
  const float* Ws  = (const float*)d_in[2];
  const float* a1  = (const float*)d_in[3];
  const float* a2  = (const float*)d_in[4];
  const float* W1  = (const float*)d_in[5];
  const float* b1  = (const float*)d_in[6];
  const float* W2  = (const float*)d_in[7];
  const float* b2  = (const float*)d_in[8];
  float* out = (float*)d_out;

  char* ws = (char*)d_ws;
  size_t off = 0;
  auto carve = [&](size_t bytes) -> char* {
    char* p = ws + off;
    off += (bytes + 255) & ~(size_t)255;
    return p;
  };
  _Float16* xh = (_Float16*)carve((size_t)BB * NN * NF * 2);          // 4 MB
  _Float16* WT = (_Float16*)carve((size_t)NL * HH * NF * 2);          // 48 KB
  _Float16* hT = (_Float16*)carve((size_t)NL * BB * HH * NN * 2);     // 6 MB
  float* f1     = (float*)carve((size_t)NL * BB * NN * 4);            // 192 KB
  float* f2     = (float*)carve((size_t)NL * BB * NN * 4);            // 192 KB
  float* pooled = (float*)carve((size_t)BB * NL * HH * 4);            // 12 KB

  prep_kernel<<<1024, 256, 0, stream>>>(x, Ws, xh, WT, pooled);
  gemm1_kernel<<<dim3(128, NL), 256, 0, stream>>>(xh, WT, a1, a2, hT, f1, f2);
  attn_kernel<<<dim3(NN / 16, BB), 256, 0, stream>>>(adj, hT, f1, f2, pooled);
  final_kernel<<<1, 256, 0, stream>>>(pooled, W1, b1, W2, b2, out);
}
